// NonLocalMeansParam_2156073583168
// MI455X (gfx1250) — matrix operand order repaired and checked by reading, not compiled, not run on hardware
//
#include <hip/hip_runtime.h>

// NLM denoise, 256x256, S=21 search & patch window.
// Box filter realized as banded 0/1 matmuls on v_wmma_f32_16x16x32_f16.
// Round 3: quad-vectorized dis-tile build (b64 packed addr map + b128 y1 loads
// + b64 f16 stores), uniform trip count via padded map; y1 term hoisted.

#define IMG   256
#define CH    65536        // 256*256
#define SRCH  21
#define PAD   10
#define NOFF  441          // 21*21
#define YT    56           // staged luminance window
#define YTS   57           // +1 to dodge LDS bank conflicts
#define DROWS 48
#define DSTR  64
#define DPAD  8            // per-wave dummy-store pad (halves)
#define DSTRW (DROWS * DSTR + DPAD)
#define NQ    324          // 36 rows x 9 column-quads
#define NQP   352          // padded to 11 * 32
#define NWAVE 4

typedef _Float16 v16h __attribute__((ext_vector_type(16)));
typedef _Float16 v8h  __attribute__((ext_vector_type(8)));
typedef _Float16 v4h  __attribute__((ext_vector_type(4)));
typedef float    v8f  __attribute__((ext_vector_type(8)));
typedef float    v4f  __attribute__((ext_vector_type(4)));

__device__ __forceinline__ int refl(int x) {
    x = (x < 0) ? -x : x;
    return (x > 255) ? (510 - x) : x;
}
__device__ __forceinline__ float clamp01(float v) {
    return fminf(fmaxf(v, 0.f), 1.f);
}
__device__ __forceinline__ int imini(int a, int b) { return a < b ? a : b; }
__device__ __forceinline__ int imaxi(int a, int b) { return a > b ? a : b; }

__global__ __launch_bounds__(128)
void nlm_wmma_kernel(const float* __restrict__ rgb,
                     const float* __restrict__ hparam,
                     float* __restrict__ out) {
    __shared__ float ytile[YT * YTS];                               // padded luminance
    __shared__ __align__(16) _Float16 disbuf[NWAVE * DSTRW];        // per-wave dis (f16)
    __shared__ __align__(32) _Float16 hmbuf[NWAVE * 16 * 64];       // per-wave Hm^T
    __shared__ __align__(8)  uint2 qmap[NQP];   // 4x13b ytile addrs + store-offset/4
    __shared__ __align__(16) float y1arr[4 * NQP];  // offset-invariant first term
    __shared__ int rowSrc[36], colSrc[36];      // reflected rgb gather indices
    __shared__ float red[NWAVE * 256];          // deterministic reduction
    __shared__ float accW[256], accR[256], accG[256], accB[256];

    const int tid  = threadIdx.x;
    const int wave = tid >> 5;
    const int lane = tid & 31;
    const int tj = blockIdx.x * 16;     // tile origin col
    const int ti = blockIdx.y * 16;     // tile origin row
    const int w0r = imini(imaxi(ti - PAD, 0), 276 - YT); // y_pad window origin
    const int w0c = imini(imaxi(tj - PAD, 0), 276 - YT);

    // Zero scratch: dis cols/rows >= 36 and Hm rows >= 48 must stay 0 so the
    // band matmuls' K-overreach multiplies against zeros.
    for (int i = tid; i < NWAVE * DSTRW; i += 128)   disbuf[i] = (_Float16)0.f;
    for (int i = tid; i < NWAVE * 16 * 64; i += 128) hmbuf[i]  = (_Float16)0.f;

    // Stage clipped-RGB luminance for y_pad rows/cols [w0*, w0*+55] (reflect pad 10).
    for (int idx = tid; idx < YT * YT; idx += 128) {
        int a = idx / YT, b = idx % YT;
        int i = refl(w0r + a - PAD), j = refl(w0c + b - PAD);
        float r  = clamp01(rgb[            i * IMG + j]);
        float g  = clamp01(rgb[    CH +    i * IMG + j]);
        float bl = clamp01(rgb[2 * CH +    i * IMG + j]);
        ytile[a * YTS + b] = 0.299f * r + 0.587f * g + 0.114f * bl;
    }

    // Per-tile offset-invariant quad map: for quad q = r*9+cq (cols 4cq..4cq+3),
    // pack the 4 reflected ytile addresses (13b each) and store-offset/4.
    // Dummy tail entries (q >= 324) target the per-wave pad slot.
    for (int q = tid; q < NQP; q += 128) {
        unsigned int a0, a1, a2, a3, so4;
        if (q < NQ) {
            int r = q / 9, c0 = (q % 9) * 4;
            int ip = refl(ti + r - PAD);
            int rowb = (ip + PAD - w0r) * YTS + PAD - w0c;
            a0 = (unsigned int)(rowb + refl(tj + c0     - PAD));
            a1 = (unsigned int)(rowb + refl(tj + c0 + 1 - PAD));
            a2 = (unsigned int)(rowb + refl(tj + c0 + 2 - PAD));
            a3 = (unsigned int)(rowb + refl(tj + c0 + 3 - PAD));
            so4 = (unsigned int)(r * 16 + (q % 9));   // (r*64 + c0) / 4
        } else {
            a0 = a1 = a2 = a3 = 1600u;                // safe mid-tile address
            so4 = (unsigned int)(DROWS * DSTR / 4);   // per-wave pad slot
        }
        uint2 m;
        m.x = a0 | (a1 << 13) | (a2 << 26);
        m.y = (a2 >> 6) | (a3 << 7) | (so4 << 20);
        qmap[q] = m;
    }
    for (int k = tid; k < 36; k += 128) {
        rowSrc[k] = refl(ti + k - PAD) * IMG;
        colSrc[k] = refl(tj + k - PAD);
    }
    __syncthreads();   // ytile/qmap ready

    // Offset-invariant first luminance term, quad-ordered for b128 loads.
    for (int q = tid; q < NQP; q += 128) {
        uint2 m = qmap[q];
        y1arr[4 * q + 0] = ytile[m.x & 0x1FFFu];
        y1arr[4 * q + 1] = ytile[(m.x >> 13) & 0x1FFFu];
        y1arr[4 * q + 2] = ytile[((m.x >> 26) | (m.y << 6)) & 0x1FFFu];
        y1arr[4 * q + 3] = ytile[(m.y >> 7) & 0x1FFFu];
    }
    __syncthreads();

    const float hv   = fmaxf(hparam[0], 0.f);
    const float invH = 1.f / (hv + 1e-8f);

    const int n16 = lane & 15;                 // N (B/C layouts), M (A layout)
    const int hiA = (lane < 16) ? 0 : 8;       // C-layout M offset
    const int hiB = (lane < 16) ? 0 : 16;      // B-layout K base

    // Constant band fragments (0/1 ones-band matrices), A and B both in contiguous K order (K = hiB + e).
    v16h bh1, bh2, va1, va2;
#pragma unroll
    for (int e = 0; e < 16; ++e) {
        int Kb = hiB + e;                                  // B layout: K = base + elem
        bh1[e] = (_Float16)((Kb >= n16 && Kb <= n16 + 20) ? 1.f : 0.f);
        bh2[e] = (_Float16)((n16 >= Kb + 12) ? 1.f : 0.f);
        int Ka = hiB + e;                                  // A layout K index (contiguous, matches B)
        va1[e] = (_Float16)((Ka >= n16 && Ka <= n16 + 20) ? 1.f : 0.f);
        va2[e] = (_Float16)((n16 >= Ka + 12) ? 1.f : 0.f);
    }

    float wac[8], rac[8], gac[8], bac[8];
#pragma unroll
    for (int r = 0; r < 8; ++r) { wac[r] = 0.f; rac[r] = 0.f; gac[r] = 0.f; bac[r] = 0.f; }

    _Float16* dbuf = &disbuf[wave * DSTRW];
    _Float16* hbuf = &hmbuf[wave * 16 * 64];   // transposed: [col*64 + row]

    // Each wave owns offsets uv = wave, wave+4, ... (independent per-wave LDS buffers).
    for (int uv = wave; uv < NOFF; uv += NWAVE) {
        const int du = uv / SRCH - PAD;
        const int dv = uv % SRCH - PAD;
        const int delta = du * YTS + dv;       // second-term ytile displacement

        // dis tile, 4 elements/iter, uniform 11-iteration trip count:
        // b64 map + b128 y1 + 4x b32 y2 gather + b64 f16 store.
        for (int q = lane; q < NQP; q += 32) {
            uint2 m = qmap[q];
            int a0 = (int)(m.x & 0x1FFFu);
            int a1 = (int)((m.x >> 13) & 0x1FFFu);
            int a2 = (int)(((m.x >> 26) | (m.y << 6)) & 0x1FFFu);
            int a3 = (int)((m.y >> 7) & 0x1FFFu);
            int so4 = (int)(m.y >> 20);
            v4f y1 = *(const v4f*)&y1arr[4 * q];
            float d0 = y1[0] - ytile[a0 + delta];
            float d1 = y1[1] - ytile[a1 + delta];
            float d2 = y1[2] - ytile[a2 + delta];
            float d3 = y1[3] - ytile[a3 + delta];
            v4h dq;
            dq[0] = (_Float16)(d0 * d0);
            dq[1] = (_Float16)(d1 * d1);
            dq[2] = (_Float16)(d2 * d2);
            dq[3] = (_Float16)(d3 * d3);
            *(v4h*)&dbuf[so4 * 4] = dq;
        }
        asm volatile("s_wait_dscnt 0" ::: "memory");   // cross-lane LDS RAW (within wave)

        // Horizontal band matmul: Hm[48x16] = X[48x64] * W  (rows >=36 are zero)
#pragma unroll
        for (int rc = 0; rc < 3; ++rc) {
            v16h a0, a1;
#pragma unroll
            for (int e = 0; e < 16; ++e) {
                int K = hiB + e;
                int row = rc * 16 + n16;
                a0[e] = dbuf[row * DSTR + K];
                a1[e] = dbuf[row * DSTR + 32 + K];
            }
            v8f hm = {};
            hm = __builtin_amdgcn_wmma_f32_16x16x32_f16(false, a0, false, bh1,
                                                        (short)0, hm, false, false);
            hm = __builtin_amdgcn_wmma_f32_16x16x32_f16(false, a1, false, bh2,
                                                        (short)0, hm, false, false);
            // contiguous b128 store of 8 halves into transposed Hm [col][row]
            v8h hh;
#pragma unroll
            for (int r = 0; r < 8; ++r) hh[r] = (_Float16)hm[r];
            *(v8h*)&hbuf[n16 * 64 + rc * 16 + hiA] = hh;
        }
        asm volatile("s_wait_dscnt 0" ::: "memory");

        // Vertical band matmul: Out = V * Hm (rows >=48 zero); contiguous 32B B loads
        const v16h vb1 = *(const v16h*)&hbuf[n16 * 64 + hiB];
        const v16h vb2 = *(const v16h*)&hbuf[n16 * 64 + 32 + hiB];
        v8f o = {};
        o = __builtin_amdgcn_wmma_f32_16x16x32_f16(false, va1, false, vb1,
                                                   (short)0, o, false, false);
        o = __builtin_amdgcn_wmma_f32_16x16x32_f16(false, va2, false, vb2,
                                                   (short)0, o, false, false);

        // Weights + weighted accumulation of reflect-padded (unclipped) RGB.
        const int sj = colSrc[n16 + dv + PAD];
#pragma unroll
        for (int r = 0; r < 8; ++r) {
            int M = r + hiA;
            float dd  = sqrtf(fmaxf(o[r], 0.f));
            float wgt = __expf(-dd * invH);
            int base = rowSrc[M + du + PAD] + sj;
            wac[r] += wgt;
            rac[r] += wgt * rgb[         base];
            gac[r] += wgt * rgb[    CH + base];
            bac[r] += wgt * rgb[2 * CH + base];
        }
    }

    // Deterministic fixed-order cross-wave reduction (no float atomics).
    for (int comp = 0; comp < 4; ++comp) {
#pragma unroll
        for (int r = 0; r < 8; ++r) {
            float v = (comp == 0) ? wac[r] : (comp == 1) ? rac[r]
                    : (comp == 2) ? gac[r] : bac[r];
            red[wave * 256 + (r + hiA) * 16 + n16] = v;
        }
        __syncthreads();
        for (int p = tid; p < 256; p += 128) {
            float s = red[p] + red[256 + p] + red[512 + p] + red[768 + p];
            if      (comp == 0) accW[p] = s;
            else if (comp == 1) accR[p] = s;
            else if (comp == 2) accG[p] = s;
            else                accB[p] = s;
        }
        __syncthreads();
    }

    for (int p = tid; p < 256; p += 128) {
        int M = p >> 4, N = p & 15;
        int pi = ti + M, pj = tj + N;
        float inv = 1.f / accW[p];
        out[         pi * IMG + pj] = clamp01(accR[p] * inv);
        out[    CH + pi * IMG + pj] = clamp01(accG[p] * inv);
        out[2 * CH + pi * IMG + pj] = clamp01(accB[p] * inv);
    }
}

extern "C" void kernel_launch(void* const* d_in, const int* in_sizes, int n_in,
                              void* d_out, int out_size, void* d_ws, size_t ws_size,
                              hipStream_t stream) {
    (void)in_sizes; (void)n_in; (void)out_size; (void)d_ws; (void)ws_size;
    const float* rgb = (const float*)d_in[0];
    const float* h   = (const float*)d_in[1];
    float* out = (float*)d_out;
    dim3 grid(IMG / 16, IMG / 16);   // 16x16 tiles
    nlm_wmma_kernel<<<grid, 128, 0, stream>>>(rgb, h, out);
}
